// Block_34806414967221
// MI455X (gfx1250) — matrix operand order repaired and checked by reading, not compiled, not run on hardware
//
#include <hip/hip_runtime.h>
#include <hip/hip_bf16.h>
#include <math.h>

// ---------------------------------------------------------------------------
// MI455X (gfx1250, wave32) implementation of the VMamba-style block.
//  - All GEMMs use V_WMMA_F32_16X16X32_F16 (f16 in, f32 accumulate),
//    32x64 macro-tile per wave (2x4 WMMA tiles, 8 accumulators) for
//    ~10.7 flops/byte of L0 traffic instead of 4 with single tiles.
//  - x and y branches folded into one 8-batch problem (tokens 0..4095 = x,
//    4096..8191 = y), matching the concatenated (a, b) output layout.
//  - Selective scan: 6144 register-resident chains, exp on TRANS pipe,
//    B/C broadcast via LDS.
// ---------------------------------------------------------------------------

typedef __attribute__((ext_vector_type(16))) _Float16 v16h;
typedef __attribute__((ext_vector_type(8)))  _Float16 v8h;
typedef __attribute__((ext_vector_type(8)))  float    v8f;

#define DIM_  384
#define DI_   768
#define NTOK  8192      // 2 branches * 4 batch * 1024 tokens
#define LSEQ  1024

union HFrag { v16h f; v8h h[2]; };

__device__ __forceinline__ float wave_sum(float v) {
  #pragma unroll
  for (int m = 16; m > 0; m >>= 1) v += __shfl_xor(v, m, 32);
  return v;
}

// --------------------------- weight conversion -----------------------------
__global__ void f32_to_f16_kernel(_Float16* __restrict__ dst,
                                  const float* __restrict__ src, int n) {
  int i = blockIdx.x * blockDim.x + threadIdx.x;
  if (i < n) dst[i] = (_Float16)src[i];
}

__global__ void padcvt_kernel(_Float16* __restrict__ dst,
                              const float* __restrict__ src,
                              int Rd, int Cd, int Rs, int Cs) {
  int idx = blockIdx.x * blockDim.x + threadIdx.x;
  if (idx >= Rd * Cd) return;
  int r = idx / Cd, c = idx % Cd;
  dst[idx] = (r < Rs && c < Cs) ? (_Float16)src[(size_t)r * Cs + c]
                                : (_Float16)0.f;
}

// --------------------------- LayerNorm (C = 384) ---------------------------
// One wave per token; srcB used for tokens >= NTOK/2 (dual-branch input).
__global__ void ln384_f16_kernel(const float* __restrict__ srcA,
                                 const float* __restrict__ srcB,
                                 const float* __restrict__ w,
                                 const float* __restrict__ b,
                                 _Float16* __restrict__ out) {
  int wid = blockIdx.x * 4 + (threadIdx.x >> 5);
  if (wid >= NTOK) return;
  int lane = threadIdx.x & 31;
  const float* src = (wid < NTOK / 2) ? srcA + (size_t)wid * DIM_
                                      : srcB + (size_t)(wid - NTOK / 2) * DIM_;
  float v[12]; float s = 0.f;
  #pragma unroll
  for (int i = 0; i < 12; ++i) { v[i] = src[lane + 32 * i]; s += v[i]; }
  s = wave_sum(s);
  float mean = s * (1.f / DIM_);
  float q = 0.f;
  #pragma unroll
  for (int i = 0; i < 12; ++i) { float d = v[i] - mean; q += d * d; }
  q = wave_sum(q);
  float rs = rsqrtf(q * (1.f / DIM_) + 1e-5f);
  #pragma unroll
  for (int i = 0; i < 12; ++i) {
    int c = lane + 32 * i;
    out[(size_t)wid * DIM_ + c] = (_Float16)((v[i] - mean) * rs * w[c] + b[c]);
  }
}

// ------------------ out_norm LN (C = 768) + SiLU(z) gate -------------------
__global__ void gatenorm_kernel(const float* __restrict__ ys,
                                const float* __restrict__ z,
                                const float* __restrict__ w,
                                const float* __restrict__ b,
                                _Float16* __restrict__ out) {
  int wid = blockIdx.x * 4 + (threadIdx.x >> 5);
  if (wid >= NTOK) return;
  int lane = threadIdx.x & 31;
  const float* row = ys + (size_t)wid * DI_;
  float v[24]; float s = 0.f;
  #pragma unroll
  for (int i = 0; i < 24; ++i) { v[i] = row[lane + 32 * i]; s += v[i]; }
  s = wave_sum(s);
  float mean = s * (1.f / DI_);
  float q = 0.f;
  #pragma unroll
  for (int i = 0; i < 24; ++i) { float d = v[i] - mean; q += d * d; }
  q = wave_sum(q);
  float rs = rsqrtf(q * (1.f / DI_) + 1e-5f);
  #pragma unroll
  for (int i = 0; i < 24; ++i) {
    int c = lane + 32 * i;
    float g = (v[i] - mean) * rs * w[c] + b[c];
    float zv = z[(size_t)wid * DI_ + c];
    g *= zv / (1.f + expf(-zv));                       // * silu(z)
    out[(size_t)wid * DI_ + c] = (_Float16)g;
  }
}

// ---------------- depthwise 3x3 'SAME' conv + bias + SiLU ------------------
__global__ void conv_silu_kernel(const float* __restrict__ xp,
                                 const float* __restrict__ cw,   // (3,3,1,768)
                                 const float* __restrict__ cb,
                                 float* __restrict__ xs,
                                 _Float16* __restrict__ xsh) {
  int idx = blockIdx.x * blockDim.x + threadIdx.x;
  if (idx >= NTOK * DI_) return;
  int c = idx % DI_; int t = idx / DI_;
  int b = t >> 10; int pos = t & 1023; int hh = pos >> 5; int ww = pos & 31;
  float acc = cb[c];
  #pragma unroll
  for (int kh = 0; kh < 3; ++kh) {
    int ih = hh + kh - 1;
    if ((unsigned)ih >= 32u) continue;
    #pragma unroll
    for (int kw = 0; kw < 3; ++kw) {
      int iw = ww + kw - 1;
      if ((unsigned)iw >= 32u) continue;
      acc += xp[(size_t)((b << 10) + (ih << 5) + iw) * DI_ + c] *
             cw[(kh * 3 + kw) * DI_ + c];
    }
  }
  float s = acc / (1.f + expf(-acc));                  // silu
  xs[idx] = s;
  xsh[idx] = (_Float16)s;
}

// -------- pack dt_r (x_dbl[:, :24]) into K=32-padded f16 A operand ---------
__global__ void dtr_pack_kernel(const float* __restrict__ xdbl,
                                _Float16* __restrict__ dtrh) {
  int idx = blockIdx.x * blockDim.x + threadIdx.x;
  if (idx >= NTOK * 32) return;
  int j = idx & 31; int t = idx >> 5;
  dtrh[idx] = (j < 24) ? (_Float16)xdbl[(size_t)t * 64 + j] : (_Float16)0.f;
}

// ------------------------- selective scan (L=1024) -------------------------
// One lane per (batch, channel); 16 SSM states in registers.
__global__ void scan_kernel(const float* __restrict__ dt,
                            const float* __restrict__ xs,
                            const float* __restrict__ xdbl,
                            const float* __restrict__ Alog,
                            const float* __restrict__ Dp,
                            float* __restrict__ ys) {
  __shared__ float bc[32];                // B (16) then C (16) for current t
  int b = blockIdx.x / 3;
  int d = (blockIdx.x % 3) * 256 + threadIdx.x;
  float Aa[16], h[16];
  #pragma unroll
  for (int n = 0; n < 16; ++n) { Aa[n] = -expf(Alog[d * 16 + n]); h[n] = 0.f; }
  float Dd = Dp[d];
  const float* dtb = dt   + (size_t)b * LSEQ * DI_;
  const float* xsb = xs   + (size_t)b * LSEQ * DI_;
  const float* xdb = xdbl + (size_t)b * LSEQ * 64;
  float* ysb       = ys   + (size_t)b * LSEQ * DI_;
  for (int t = 0; t < LSEQ; ++t) {
    __syncthreads();
    if (threadIdx.x < 32) bc[threadIdx.x] = xdb[(size_t)t * 64 + 24 + threadIdx.x];
    __syncthreads();
    float dtv = dtb[(size_t)t * DI_ + d];
    float xv  = xsb[(size_t)t * DI_ + d];
    float out = 0.f;
    #pragma unroll
    for (int n = 0; n < 16; ++n) {
      float dA = __expf(dtv * Aa[n]);                  // v_exp_f32 (TRANS)
      h[n] = dA * h[n] + (dtv * bc[n]) * xv;           // dA*h + dt*B*u
      out += h[n] * bc[16 + n];                        // h . C
    }
    ysb[(size_t)t * DI_ + d] = out + xv * Dd;
  }
}

// ------------------------------ WMMA GEMM ----------------------------------
// C[M,N] = A[M,K](f16,row-major) * W[N,K]^T(f16,row-major), f32 accumulate.
// 32x64 macro-tile per wave32: 2 M-tiles x 4 N-tiles, 8 v8f accumulators.
// Fragment layouts as loaded here (contiguous K, not the ISA table order):
//   A 16x32 f16: lane row = lane&15; VGPR v = K 16*hi + {2v, 2v+1}
//   B 32x16 f16: lane col = lane&15; VGPR v = K 16*hi + {2v, 2v+1}
//   C/D f32:     row = r + 8*hi, col = lane&15
enum { GM_XZ = 0, GM_PLAIN = 1, GM_SOFTPLUS = 2, GM_GELU = 3,
       GM_RESID2 = 4, GM_BIAS_RESID = 5 };

__global__ void gemm_wmma_kernel(const _Float16* __restrict__ A,
                                 const _Float16* __restrict__ B,
                                 int M, int N, int K, int mode,
                                 const float* __restrict__ bias,
                                 float* __restrict__ out0,
                                 float* __restrict__ out1,
                                 _Float16* __restrict__ outh,
                                 const float* __restrict__ resid,
                                 const float* __restrict__ rx,
                                 const float* __restrict__ ry) {
  int wid    = blockIdx.x * (blockDim.x >> 5) + (threadIdx.x >> 5);
  int tilesN = N >> 6;                         // 64-wide macro tiles
  int tiles  = (M >> 5) * tilesN;              // 32-tall macro tiles
  if (wid >= tiles) return;                    // wave-uniform
  int tm = wid / tilesN, tn = wid % tilesN;
  int lane = threadIdx.x & 31;
  int mr = lane & 15, hi = lane >> 4;
  const _Float16* Arow = A + (size_t)(tm * 32 + mr) * K;   // M-tile 0 row
  const _Float16* Brow = B + (size_t)(tn * 64 + mr) * K;   // N-tile 0 row
  size_t aStep = (size_t)16 * K;               // next M-tile / N-tile
  v8f acc[2][4] = {};
  for (int k0 = 0; k0 < K; k0 += 32) {
    HFrag a[2], bf[4];
    #pragma unroll
    for (int i = 0; i < 2; ++i) {
      a[i].h[0] = *(const v8h*)(Arow + i * aStep + k0 + 16 * hi);
      a[i].h[1] = *(const v8h*)(Arow + i * aStep + k0 + 16 * hi + 8);
      __builtin_prefetch(Arow + i * aStep + k0 + 32, 0, 2);
    }
    #pragma unroll
    for (int j = 0; j < 4; ++j) {
      bf[j].h[0] = *(const v8h*)(Brow + j * aStep + k0 + 16 * hi);
      bf[j].h[1] = *(const v8h*)(Brow + j * aStep + k0 + 16 * hi + 8);
    }
    #pragma unroll
    for (int i = 0; i < 2; ++i)
      #pragma unroll
      for (int j = 0; j < 4; ++j)
        acc[i][j] = __builtin_amdgcn_wmma_f32_16x16x32_f16(
            false, a[i].f, false, bf[j].f, (short)0, acc[i][j], false, false);
  }
  #pragma unroll
  for (int i = 0; i < 2; ++i) {
    #pragma unroll
    for (int j = 0; j < 4; ++j) {
      int n = tn * 64 + j * 16 + mr;
      #pragma unroll
      for (int r = 0; r < 8; ++r) {
        int m = tm * 32 + i * 16 + r + 8 * hi;
        float v = acc[i][j][r];
        switch (mode) {
          case GM_XZ: {                            // in_proj: split xp | z
            v += bias[n];
            if (n < DI_) out0[(size_t)m * DI_ + n] = v;
            else         out1[(size_t)m * DI_ + (n - DI_)] = v;
          } break;
          case GM_PLAIN:
            out0[(size_t)m * N + n] = v;
            break;
          case GM_SOFTPLUS: {                      // dt = softplus(v + b)
            v += bias[n];
            v = fmaxf(v, 0.f) + log1pf(expf(-fabsf(v)));
            out0[(size_t)m * N + n] = v;
          } break;
          case GM_GELU: {                          // exact gelu -> f16
            v += bias[n];
            v = 0.5f * v * (1.f + erff(v * 0.70710678118f));
            outh[(size_t)m * N + n] = (_Float16)v;
          } break;
          case GM_RESID2: {                        // + x or y residual
            v += (m < NTOK / 2) ? rx[(size_t)m * N + n]
                                : ry[(size_t)(m - NTOK / 2) * N + n];
            out0[(size_t)m * N + n] = v;
          } break;
          case GM_BIAS_RESID: {                    // + bias + residual
            v += bias[n] + resid[(size_t)m * N + n];
            out0[(size_t)m * N + n] = v;
          } break;
        }
      }
    }
  }
}

// ------------------------------- launcher ----------------------------------
extern "C" void kernel_launch(void* const* d_in, const int* in_sizes, int n_in,
                              void* d_out, int out_size, void* d_ws,
                              size_t ws_size, hipStream_t stream) {
  const float* X    = (const float*)d_in[0];
  const float* Y    = (const float*)d_in[1];
  const float* ln1w = (const float*)d_in[2];
  const float* ln1b = (const float*)d_in[3];
  const float* Wi   = (const float*)d_in[4];
  const float* bi   = (const float*)d_in[5];
  const float* cw   = (const float*)d_in[6];
  const float* cb   = (const float*)d_in[7];
  const float* Wx   = (const float*)d_in[8];
  const float* Wdt  = (const float*)d_in[9];
  const float* bdt  = (const float*)d_in[10];
  const float* Alog = (const float*)d_in[11];
  const float* Dp   = (const float*)d_in[12];
  const float* onw  = (const float*)d_in[13];
  const float* onb  = (const float*)d_in[14];
  const float* Wo   = (const float*)d_in[15];
  const float* ln2w = (const float*)d_in[16];
  const float* ln2b = (const float*)d_in[17];
  const float* Wf1  = (const float*)d_in[18];
  const float* bf1  = (const float*)d_in[19];
  const float* Wf2  = (const float*)d_in[20];
  const float* bf2  = (const float*)d_in[21];
  (void)in_sizes; (void)n_in; (void)out_size; (void)ws_size;

  char* Wb = (char*)d_ws;
  size_t off = 0;
  auto take = [&](size_t bytes) -> size_t {
    size_t r = off; off = (off + bytes + 255) & ~(size_t)255; return r;
  };
  float*    xp   = (float*)(Wb + take((size_t)NTOK * DI_ * 4));
  float*    z    = (float*)(Wb + take((size_t)NTOK * DI_ * 4));
  float*    xs   = (float*)(Wb + take((size_t)NTOK * DI_ * 4));
  float*    ys   = (float*)(Wb + take((size_t)NTOK * DI_ * 4));
  float*    xdbl = (float*)(Wb + take((size_t)NTOK * 64 * 4));
  float*    aRes = (float*)(Wb + take((size_t)NTOK * DIM_ * 4));
  _Float16* A0h  = (_Float16*)(Wb + take((size_t)NTOK * DIM_ * 2));
  _Float16* xsh  = (_Float16*)(Wb + take((size_t)NTOK * DI_ * 2));
  _Float16* dtrh = (_Float16*)(Wb + take((size_t)NTOK * 32 * 2));
  _Float16* wih  = (_Float16*)(Wb + take((size_t)1536 * 384 * 2));
  _Float16* wxh  = (_Float16*)(Wb + take((size_t)64 * 768 * 2));
  _Float16* wdth = (_Float16*)(Wb + take((size_t)768 * 32 * 2));
  _Float16* woh  = (_Float16*)(Wb + take((size_t)384 * 768 * 2));
  _Float16* wf1h = (_Float16*)(Wb + take((size_t)192 * 384 * 2));
  _Float16* wf2h = (_Float16*)(Wb + take((size_t)384 * 192 * 2));
  // Aliases into regions that are dead by the time these are written:
  float*    dtf  = xp;                 // dt overwrites xp (dead after conv)
  _Float16* gh   = (_Float16*)xs;      // gate out overwrites xs (dead after scan)
  _Float16* alnh = A0h;                // ln2 out reuses ln1 out
  _Float16* h1h  = xsh;                // MLP hidden reuses xs_h
  float*    outF = (float*)d_out;

  auto gemm = [&](const _Float16* Ap, const _Float16* Bp, int M, int N, int K,
                  int mode, const float* bias, float* o0, float* o1,
                  _Float16* oh, const float* resid, const float* rx,
                  const float* ry) {
    int waves = (M / 32) * (N / 64);
    gemm_wmma_kernel<<<(waves + 3) / 4, 128, 0, stream>>>(
        Ap, Bp, M, N, K, mode, bias, o0, o1, oh, resid, rx, ry);
  };

  // Weight conversion (f32 -> f16, with K/N zero padding where needed)
  f32_to_f16_kernel<<<(1536 * 384 + 255) / 256, 256, 0, stream>>>(wih, Wi, 1536 * 384);
  padcvt_kernel<<<(64 * 768 + 255) / 256, 256, 0, stream>>>(wxh, Wx, 64, 768, 56, 768);
  padcvt_kernel<<<(768 * 32 + 255) / 256, 256, 0, stream>>>(wdth, Wdt, 768, 32, 768, 24);
  f32_to_f16_kernel<<<(384 * 768 + 255) / 256, 256, 0, stream>>>(woh, Wo, 384 * 768);
  f32_to_f16_kernel<<<(192 * 384 + 255) / 256, 256, 0, stream>>>(wf1h, Wf1, 192 * 384);
  f32_to_f16_kernel<<<(384 * 192 + 255) / 256, 256, 0, stream>>>(wf2h, Wf2, 384 * 192);

  // Mix path (both branches folded, tokens 0..8191)
  ln384_f16_kernel<<<NTOK / 4, 128, 0, stream>>>(X, Y, ln1w, ln1b, A0h);
  gemm(A0h, wih, NTOK, 1536, 384, GM_XZ, bi, xp, z, nullptr, nullptr, nullptr, nullptr);
  conv_silu_kernel<<<(NTOK * DI_ + 255) / 256, 256, 0, stream>>>(xp, cw, cb, xs, xsh);
  gemm(xsh, wxh, NTOK, 64, 768, GM_PLAIN, nullptr, xdbl, nullptr, nullptr, nullptr, nullptr, nullptr);
  dtr_pack_kernel<<<(NTOK * 32 + 255) / 256, 256, 0, stream>>>(xdbl, dtrh);
  gemm(dtrh, wdth, NTOK, 768, 32, GM_SOFTPLUS, bdt, dtf, nullptr, nullptr, nullptr, nullptr, nullptr);
  scan_kernel<<<24, 256, 0, stream>>>(dtf, xs, xdbl, Alog, Dp, ys);
  gatenorm_kernel<<<NTOK / 4, 128, 0, stream>>>(ys, z, onw, onb, gh);
  gemm(gh, woh, NTOK, 384, 768, GM_RESID2, nullptr, aRes, nullptr, nullptr, nullptr, X, Y);

  // MLP path with residual; final GEMM writes d_out = concat(a, b)
  ln384_f16_kernel<<<NTOK / 4, 128, 0, stream>>>(
      aRes, aRes + (size_t)(NTOK / 2) * DIM_, ln2w, ln2b, alnh);
  gemm(alnh, wf1h, NTOK, 192, 384, GM_GELU, bf1, nullptr, nullptr, h1h, nullptr, nullptr, nullptr);
  gemm(h1h, wf2h, NTOK, 384, 192, GM_BIAS_RESID, bf2, outF, nullptr, nullptr, aRes, nullptr, nullptr);
}
